// SNC_conv_69200513073410
// MI455X (gfx1250) — matrix operand order repaired and checked by reading, not compiled, not run on hardware
//
#include <hip/hip_runtime.h>
#include <hip/hip_fp16.h>

typedef __attribute__((ext_vector_type(16))) _Float16 v16h;
typedef __attribute__((ext_vector_type(8)))  float    v8f;

#define EPSF 1e-20f
#define HDIM 192
#define WDIM 192
#define HW   (192*192)

// d_ws layout (bytes)
#define OFF_PACKAE 0            // 8*36*32*16 f16 = 294912 B  (w_conv_e, A-frag order)
#define OFF_PACKAD 294912       // 2*9*32*16  f16 = 18432 B   (w_conv_d, A-frag order)
#define OFF_WPOW   313344       // 32*2*4 f32 = 1024 B
#define OFF_WPROP  314368       // 32*4  f32 = 512 B
#define OFF_CEIN   1048576      // 2*HW*128 f16 = 18874368 B  (channel-last)
#define OFF_ECEIN  19922944     // 2*HW*128 f16 = 18874368 B

__device__ inline float softplusf(float v) { return (v > 20.f) ? v : log1pf(expf(v)); }

// ---------------------------------------------------------------------------
// Kernel 0: weight prep + A-fragment packing (tiny; one block)
// ---------------------------------------------------------------------------
__global__ __launch_bounds__(256) void prep_weights(
    const float* __restrict__ Wcd,  const float* __restrict__ Wsd,
    const float* __restrict__ Wce,  const float* __restrict__ Wse0,
    const float* __restrict__ Wse1, const float* __restrict__ Wse3,
    const float* __restrict__ Wdir, const float* __restrict__ Wpow,
    const float* __restrict__ Wprop,
    _Float16* __restrict__ packAe, _Float16* __restrict__ packAd,
    float* __restrict__ wpow_t, float* __restrict__ wprop_t)
{
    __shared__ float s_wcd[1024];   // [o][i]
    __shared__ float s_wce[1024];   // [o][i]
    __shared__ float s_wsd[288];    // [n][k]   k = h*3+w (col2 = col0)
    __shared__ float s_wdir[512];   // [(p*32+n)*4 + d]
    __shared__ float s_wse[1152];   // [(n*4+d)*9 + k]
    const int tid = threadIdx.x;

    for (int i = tid; i < 1024; i += 256) {
        s_wcd[i] = softplusf(Wcd[i]);
        s_wce[i] = softplusf(Wce[i]);
    }
    for (int i = tid; i < 288; i += 256) {
        int n = i / 9, k = i % 9, h = k / 3, w = k % 3;
        int wc = (w == 2) ? 0 : w;                  // mirrored col
        s_wsd[i] = softplusf(Wsd[(n * 3 + h) * 2 + wc]);
    }
    for (int i = tid; i < 512; i += 256) {
        int d = i & 3, n = (i >> 2) & 31, p = i >> 7;
        const int dmap[4][4] = {{0,1,2,3},{4,5,4,6},{2,1,0,3},{7,8,7,9}};
        s_wdir[i] = softplusf(Wdir[n * 10 + dmap[p][d]]);
    }
    for (int i = tid; i < 1152; i += 256) {
        int k = i % 9, d = (i / 9) & 3, n = i / 36;
        int h = k / 3, w = k % 3;
        float v;
        if (d == 0)      v = softplusf(Wse0[(n * 3 + h) * 3 + w]);
        else if (d == 2) v = softplusf(Wse0[(n * 3 + h) * 3 + (2 - w)]);
        else {
            const float* src = (d == 1) ? Wse1 : Wse3;
            int wc = (w == 2) ? 0 : w;
            v = softplusf(src[(n * 3 + h) * 2 + wc]);
        }
        s_wse[i] = v;
    }
    __syncthreads();

    if (tid < 32) { // w_channel_d rows
        float s = 0.f; for (int i = 0; i < 32; ++i) s += s_wcd[tid * 32 + i];
        float inv = 1.f / s; for (int i = 0; i < 32; ++i) s_wcd[tid * 32 + i] *= inv;
    }
    if (tid < 32) { // w_channel_e rows
        float s = 0.f; for (int i = 0; i < 32; ++i) s += s_wce[tid * 32 + i];
        float inv = 1.f / s; for (int i = 0; i < 32; ++i) s_wce[tid * 32 + i] *= inv;
    }
    if (tid < 32) { // w_spatial_d per n over 9
        float s = 0.f; for (int k = 0; k < 9; ++k) s += s_wsd[tid * 9 + k];
        float inv = 1.f / s; for (int k = 0; k < 9; ++k) s_wsd[tid * 9 + k] *= inv;
    }
    if (tid < 128) { // w_dir per (p,n) over 4
        float s = 0.f; for (int d = 0; d < 4; ++d) s += s_wdir[tid * 4 + d];
        float inv = 1.f / s; for (int d = 0; d < 4; ++d) s_wdir[tid * 4 + d] *= inv;
    }
    if (tid < 128) { // w_spatial_e per (n,d) over 9
        float s = 0.f; for (int k = 0; k < 9; ++k) s += s_wse[tid * 9 + k];
        float inv = 1.f / s; for (int k = 0; k < 9; ++k) s_wse[tid * 9 + k] *= inv;
    }
    __syncthreads();

    // Pack w_conv_e into WMMA A-fragment order.
    // K ordering: k_global = tap*128 + ic ; step s = tap*4 + chunk (chunk of 32 ch)
    // Per 16x32 A-frag: lane<16: K=j ; lane>=16: K=j+16
    for (int e = tid; e < 8 * 36 * 32 * 16; e += 256) {
        int j = e & 15, lane = (e >> 4) & 31, s = (e >> 9) % 36, T = e / (512 * 36);
        int t = s >> 2, c = s & 3;
        int klocal = (lane >> 4) * 16 + j;
        int ic = 32 * c + klocal;
        int oc = 16 * T + (lane & 15);
        int o = oc >> 2, p = oc & 3, i = ic >> 2, dd = ic & 3;
        float wgt = s_wce[o * 32 + i] * s_wdir[(p * 32 + i) * 4 + dd] * s_wse[(i * 4 + dd) * 9 + t];
        packAe[e] = (_Float16)wgt;
    }
    // Pack w_conv_d (K = tap*32 + ic, 9 steps, 2 M-tiles)
    for (int e = tid; e < 2 * 9 * 32 * 16; e += 256) {
        int j = e & 15, lane = (e >> 4) & 31, t = (e >> 9) % 9, T = e / 4608;
        int klocal = (lane >> 4) * 16 + j;
        int ic = klocal, oc = 16 * T + (lane & 15);
        packAd[e] = (_Float16)(s_wcd[oc * 32 + ic] * s_wsd[ic * 9 + t]);
    }
    if (tid < 32) {
        float p5[5];
        for (int j = 0; j < 5; ++j) p5[j] = softplusf(Wpow[tid * 5 + j]);
        // w_pow[n][comp][dir]: dir0:(p0,p1) dir1:(p2,p3) dir2:(p1,p0) dir3:(p4,p4)
        wpow_t[tid * 8 + 0] = p5[0]; wpow_t[tid * 8 + 1] = p5[2];
        wpow_t[tid * 8 + 2] = p5[1]; wpow_t[tid * 8 + 3] = p5[4];
        wpow_t[tid * 8 + 4] = p5[1]; wpow_t[tid * 8 + 5] = p5[3];
        wpow_t[tid * 8 + 6] = p5[0]; wpow_t[tid * 8 + 7] = p5[4];
        float s0 = 1.f / (1.f + expf(-Wprop[tid * 3 + 0]));
        float s1 = 1.f / (1.f + expf(-Wprop[tid * 3 + 1]));
        float s2 = 1.f / (1.f + expf(-Wprop[tid * 3 + 2]));
        wprop_t[tid * 4 + 0] = s1; wprop_t[tid * 4 + 1] = s0;
        wprop_t[tid * 4 + 2] = s1; wprop_t[tid * 4 + 3] = s2;
    }
}

// ---------------------------------------------------------------------------
// Kernel 1: edge/confidence prep -> ce_in/ece_in, channel-last f16
// ---------------------------------------------------------------------------
__global__ __launch_bounds__(256) void prep_inputs(
    const float* __restrict__ x, const float* __restrict__ ece, const float* __restrict__ ce,
    const float* __restrict__ wpow_t, const float* __restrict__ wprop_t,
    _Float16* __restrict__ cein, _Float16* __restrict__ ecein)
{
    int idx = blockIdx.x * 256 + threadIdx.x;   // = pix*128 + c
    int c   = idx & 127;
    int pix = idx >> 7;                          // b*HW + h*W + w
    int w = pix % WDIM;
    int h = (pix / WDIM) % HDIM;
    int b = pix / HW;
    int n = c >> 2, dir = c & 3;
    const float* dcd = x + (size_t)(b * 32 + n) * HW;
    const float* cd  = x + (size_t)((b + 2) * 32 + n) * HW;

    int dy = (dir == 3) ? 0 : 1;
    int dx = (dir == 0) ? 1 : (dir == 1) ? 0 : -1;
    int ha = h - dy, wa = w - dx, hb = h + dy, wb = w + dx;
    float cda = 0.f, cdb = 0.f, da = 0.f, db = 0.f;
    if (ha >= 0 && ha < HDIM && wa >= 0 && wa < WDIM) {
        float cv = cd[ha * WDIM + wa]; cda = cv; da = dcd[ha * WDIM + wa] / (cv + EPSF);
    }
    if (hb >= 0 && hb < HDIM && wb >= 0 && wb < WDIM) {
        float cv = cd[hb * WDIM + wb]; cdb = cv; db = dcd[hb * WDIM + wb] / (cv + EPSF);
    }
    float ce_new = cda * cdb;
    float rf = fminf(fmaxf(da / (db + EPSF), EPSF), 1.f);
    float rb = fminf(fmaxf(db / (da + EPSF), EPSF), 1.f);
    float p0 = wpow_t[n * 8 + dir], p1 = wpow_t[n * 8 + 4 + dir];
    float en = fminf(expf(p0 * logf(rf)), expf(p1 * logf(rb)));   // pow, base>0
    en = fminf(fmaxf(en, 0.f), 1.f);
    float wp = wprop_t[n * 4 + dir];
    size_t ci = (size_t)((b * 32 + n) * 4 + dir) * HW + h * WDIM + w;
    float cev = ce[ci], ecev = ece[ci];
    cein[idx]  = (_Float16)(cev * wp + ce_new * (1.f - wp));
    ecein[idx] = (_Float16)(ecev * wp + en * ce_new * (1.f - wp));
}

// ---------------------------------------------------------------------------
// Kernel 2: dual 128->128 3x3 conv via WMMA f16 (ce_out, ece_out)
// Block = 8 waves; wave w -> out-channel tile w; 16-pixel row tile per block.
// ---------------------------------------------------------------------------
__global__ __launch_bounds__(256) void conv_e_kernel(
    const _Float16* __restrict__ cein, const _Float16* __restrict__ ecein,
    const _Float16* __restrict__ packA,
    float* __restrict__ ece_out, float* __restrict__ ce_out)
{
    __shared__ __align__(16) _Float16 s_ce[3 * 18 * 128];
    __shared__ __align__(16) _Float16 s_ece[3 * 18 * 128];
    int blk  = blockIdx.x;                 // b*H*12 + row*12 + ct
    int ct   = blk % 12;
    int row  = (blk / 12) % HDIM;
    int b    = blk / (12 * HDIM);
    int col0 = ct * 16;
    int tid  = threadIdx.x;

    // Stage 3x18 pixel columns x 128 channels of both inputs (16B chunks)
    for (int chunk = tid; chunk < 864; chunk += 256) {
        int cc   = chunk & 15;            // 8-channel chunk
        int pc   = chunk >> 4;            // 0..53
        int pcol = pc % 18, prow = pc / 18;
        int gh = row + prow - 1, gw = col0 + pcol - 1;
        float4 v0 = make_float4(0.f, 0.f, 0.f, 0.f), v1 = v0;
        if (gh >= 0 && gh < HDIM && gw >= 0 && gw < WDIM) {
            size_t base = ((size_t)(b * HW + gh * WDIM + gw) * 128) + cc * 8;
            v0 = *(const float4*)(cein + base);
            v1 = *(const float4*)(ecein + base);
        }
        *(float4*)(&s_ce[pc * 128 + cc * 8])  = v0;
        *(float4*)(&s_ece[pc * 128 + cc * 8]) = v1;
    }
    __syncthreads();

    int wave = tid >> 5, lane = tid & 31;
    int ln = lane & 15, hh = lane >> 4;    // pixel within tile / K-half
    v8f acc_ce = {}, acc_ece = {};

    for (int t = 0; t < 9; ++t) {
        int dy = t / 3, dx = t % 3;
        for (int c = 0; c < 4; ++c) {
            int s = t * 4 + c;
            v16h a = *(const v16h*)(packA + ((wave * 36 + s) * 32 + lane) * 16);
            const _Float16* bp = &s_ce[(dy * 18 + ln + dx) * 128 + c * 32 + hh * 16];
            const _Float16* ep = &s_ece[(dy * 18 + ln + dx) * 128 + c * 32 + hh * 16];
            v16h bce  = *(const v16h*)bp;
            v16h bece = *(const v16h*)ep;
            acc_ce  = __builtin_amdgcn_wmma_f32_16x16x32_f16(false, a, false, bce,  (short)0, acc_ce,  false, false);
            acc_ece = __builtin_amdgcn_wmma_f32_16x16x32_f16(false, a, false, bece, (short)0, acc_ece, false, false);
        }
    }
    // D: VGPR r -> oc = wave*16 + hh*8 + r ; lane&15 -> pixel
    int gw = col0 + ln;
    for (int r = 0; r < 8; ++r) {
        int oc = wave * 16 + hh * 8 + r;
        size_t o = (size_t)(b * 128 + oc) * HW + row * WDIM + gw;
        ce_out[o]  = acc_ce[r];
        ece_out[o] = acc_ece[r];
    }
}

// ---------------------------------------------------------------------------
// Kernel 3: edge-gated 32->32 3x3 einsum-conv for dcd_out / cd_out via WMMA
// Block = 4 waves: (dcd,cd) x (M-tile 0,1); 16-pixel row tile.
// ---------------------------------------------------------------------------
__global__ __launch_bounds__(128) void final_conv(
    const float* __restrict__ x, const _Float16* __restrict__ packAd,
    const float* __restrict__ ece_out, const float* __restrict__ ce_out,
    float* __restrict__ dout)
{
    __shared__ __align__(16) _Float16 s_dcd[3 * 18 * 32];
    __shared__ __align__(16) _Float16 s_cd [3 * 18 * 32];
    __shared__ __align__(16) _Float16 s_e  [4 * 16 * 32];   // [dir][px][ch]
    int blk  = blockIdx.x;
    int ct   = blk % 12;
    int row  = (blk / 12) % HDIM;
    int b    = blk / (12 * HDIM);
    int col0 = ct * 16;
    int tid  = threadIdx.x;

    for (int i = tid; i < 1728; i += 128) {
        int ch = i & 31, pc = i >> 5;
        int pcol = pc % 18, prow = pc / 18;
        int gh = row + prow - 1, gw = col0 + pcol - 1;
        float dv = 0.f, cv = 0.f;
        if (gh >= 0 && gh < HDIM && gw >= 0 && gw < WDIM) {
            size_t gi = (size_t)(b * 32 + ch) * HW + gh * WDIM + gw;
            dv = x[gi];
            cv = x[gi + (size_t)2 * 32 * HW];
        }
        s_dcd[pc * 32 + ch] = (_Float16)dv;
        s_cd [pc * 32 + ch] = (_Float16)cv;
    }
    for (int i = tid; i < 2048; i += 128) {
        int ch = i & 31, px = (i >> 5) & 15, dir = i >> 9;
        size_t gi = (size_t)(b * 128 + ch * 4 + dir) * HW + row * WDIM + col0 + px;
        float e = ece_out[gi] / (ce_out[gi] + EPSF);
        e = fminf(fmaxf(e, EPSF), 1.f);
        s_e[dir * 512 + px * 32 + ch] = (_Float16)e;
    }
    __syncthreads();

    int wave = tid >> 5, lane = tid & 31;
    int ln = lane & 15, hh = lane >> 4;
    int mt = wave & 1;
    const _Float16* s_src = (wave < 2) ? s_dcd : s_cd;
    const int tapdir[9] = {0, 1, 2, 3, -1, 3, 2, 1, 0};
    v8f acc = {};

    for (int t = 0; t < 9; ++t) {
        int dy = t / 3, dx = t % 3;
        v16h a = *(const v16h*)(packAd + ((mt * 9 + t) * 32 + lane) * 16);
        const _Float16* dp = &s_src[(dy * 18 + ln + dx) * 32 + hh * 16];
        v16h bvec;
        if (t == 4) {
            bvec = *(const v16h*)dp;                         // center tap: e=1
        } else {
            v16h dv = *(const v16h*)dp;
            v16h ev = *(const v16h*)(&s_e[tapdir[t] * 512 + ln * 32 + hh * 16]);
            for (int j = 0; j < 16; ++j)
                bvec[j] = (_Float16)((float)dv[j] * (float)ev[j]);
        }
        acc = __builtin_amdgcn_wmma_f32_16x16x32_f16(false, a, false, bvec, (short)0, acc, false, false);
    }
    int ob = (wave < 2) ? b : b + 2;                         // concat([dcd_out, cd_out], axis=0)
    for (int r = 0; r < 8; ++r) {
        int oc = mt * 16 + hh * 8 + r;
        dout[(size_t)(ob * 32 + oc) * HW + row * WDIM + col0 + ln] = acc[r];
    }
}

// ---------------------------------------------------------------------------
extern "C" void kernel_launch(void* const* d_in, const int* in_sizes, int n_in,
                              void* d_out, int out_size, void* d_ws, size_t ws_size,
                              hipStream_t stream) {
    const float* x     = (const float*)d_in[0];
    const float* ece   = (const float*)d_in[1];
    const float* ce    = (const float*)d_in[2];
    const float* Wcd   = (const float*)d_in[3];
    const float* Wsd   = (const float*)d_in[4];
    const float* Wce   = (const float*)d_in[5];
    const float* Wse0  = (const float*)d_in[6];
    const float* Wse1  = (const float*)d_in[7];
    const float* Wse3  = (const float*)d_in[8];
    const float* Wdir  = (const float*)d_in[9];
    const float* Wpow  = (const float*)d_in[10];
    const float* Wprop = (const float*)d_in[11];

    char* ws = (char*)d_ws;
    _Float16* packAe  = (_Float16*)(ws + OFF_PACKAE);
    _Float16* packAd  = (_Float16*)(ws + OFF_PACKAD);
    float*    wpow_t  = (float*)(ws + OFF_WPOW);
    float*    wprop_t = (float*)(ws + OFF_WPROP);
    _Float16* cein    = (_Float16*)(ws + OFF_CEIN);
    _Float16* ecein   = (_Float16*)(ws + OFF_ECEIN);

    float* out    = (float*)d_out;
    float* eceout = out + (size_t)4 * 32 * HW;                 // after concat output
    float* ceout  = eceout + (size_t)2 * 128 * HW;

    prep_weights<<<1, 256, 0, stream>>>(Wcd, Wsd, Wce, Wse0, Wse1, Wse3, Wdir, Wpow, Wprop,
                                        packAe, packAd, wpow_t, wprop_t);
    prep_inputs<<<(2 * HW * 128) / 256, 256, 0, stream>>>(x, ece, ce, wpow_t, wprop_t, cein, ecein);
    conv_e_kernel<<<2 * HDIM * 12, 256, 0, stream>>>(cein, ecein, packAe, eceout, ceout);
    final_conv<<<2 * HDIM * 12, 128, 0, stream>>>(x, packAd, eceout, ceout, out);
}
